// DeformableDETREncoder_70592082477511
// MI455X (gfx1250) — compile-verified, matrix operand order checked by reading, not run on hardware
//
#include <hip/hip_runtime.h>
#include <hip/hip_bf16.h>

#define D_MODEL   256
#define N_HEADS   8
#define HEAD_DIM  32
#define D_FFN     1024
#define NUM_LAYERS 6
#define BATCH     2
#define QLEN      13294            // 100*100 + 50*50 + 25*25 + 13*13
#define MROWS     (BATCH * QLEN)   // 26588
#define MPAD      26624            // 208 * 128

typedef __attribute__((ext_vector_type(16))) _Float16 v16h;
typedef __attribute__((ext_vector_type(8)))  _Float16 v8h;
typedef __attribute__((ext_vector_type(8)))  float    v8f;

// ---------------------------------------------------------------------------
// Weight convert + transpose: src is f32 [K][N] row-major, dst is f16 [N][K].
// ---------------------------------------------------------------------------
__global__ void k_transcvt(const float* __restrict__ src, _Float16* __restrict__ dst,
                           int K, int N) {
    int i = blockIdx.x * 256 + threadIdx.x;
    if (i >= K * N) return;
    int k = i / N, n = i % N;
    dst[(size_t)n * K + k] = (_Float16)src[i];
}

// out -> x16 (f16) and out+pos -> q16 (f16)
__global__ void k_addcvt(const float* __restrict__ out, const float* __restrict__ pos,
                         _Float16* __restrict__ q16, _Float16* __restrict__ x16, int n) {
    int i = blockIdx.x * 256 + threadIdx.x;
    if (i >= n) return;
    float o = out[i];
    x16[i] = (_Float16)o;
    q16[i] = (_Float16)(o + pos[i]);
}

__global__ void k_cvt(const float* __restrict__ src, _Float16* __restrict__ dst, int n) {
    int i = blockIdx.x * 256 + threadIdx.x;
    if (i >= n) return;
    dst[i] = (_Float16)src[i];
}

// ---------------------------------------------------------------------------
// WMMA GEMM: C[M,N] = A[M,K] (f16, row-major) * Bt[N,K] (f16, weights pre-
// transposed, so B fragments load contiguously) + bias[N].
// Block = 128 threads (4 waves); wave w computes a 32x64 tile:
//   rows blockIdx.x*128 + w*32 .. +32, cols blockIdx.y*64 .. +64.
// Per K-step: 2 A fragments + 4 B fragments (distinct registers, loaded up
// front -> one s_clause, incremental s_wait_loadcnt) feed 8 back-to-back
// v_wmma_f32_16x16x32_f16.  B fragments are reused across both row tiles.
// Epilogue: uniform fast path (row tile fully in-range -> unpredicated
// stores, row stride folded into immediate offsets); predicated path only
// for the single tail row-block.
// EPI = 0: store f32 (Cf).  EPI = 1: bias + ReLU, store f16 (Ch).
// ---------------------------------------------------------------------------
template <int EPI>
__global__ __launch_bounds__(128) void k_gemm(const _Float16* __restrict__ A,
                                              const _Float16* __restrict__ Bt,
                                              const float* __restrict__ bias,
                                              float* __restrict__ Cf,
                                              _Float16* __restrict__ Ch,
                                              int M, int N, int K) {
    const int lane = threadIdx.x & 31;
    const int wv   = threadIdx.x >> 5;
    const int row0 = blockIdx.x * 128 + wv * 32;
    const int col0 = blockIdx.y * 64;
    const int hh   = lane >> 4;     // lane half: selects K-subgroup
    const int l16  = lane & 15;

    int m0 = row0 + l16;       if (m0 > M - 1) m0 = M - 1;   // clamp: EXEC stays full
    int m1 = row0 + 16 + l16;  if (m1 > M - 1) m1 = M - 1;
    const _Float16* __restrict__ arow0 = A + (size_t)m0 * K;
    const _Float16* __restrict__ arow1 = A + (size_t)m1 * K;

    v8f acc[2][4] = {};

    for (int k0 = 0; k0 < K; k0 += 32) {
        // CDNA5 16-bit A layout: elems 0..7 = k0+hh*8 .. +8, elems 8..15 = k0+16+hh*8 .. +8
        v16h a0, a1, bf[4];
        *((v8h*)&a0 + 0) = *(const v8h*)(arow0 + k0 + hh * 8);
        *((v8h*)&a0 + 1) = *(const v8h*)(arow0 + k0 + 16 + hh * 8);
        *((v8h*)&a1 + 0) = *(const v8h*)(arow1 + k0 + hh * 8);
        *((v8h*)&a1 + 1) = *(const v8h*)(arow1 + k0 + 16 + hh * 8);
#pragma unroll
        for (int t = 0; t < 4; ++t) {
            const _Float16* __restrict__ brow = Bt + (size_t)(col0 + t * 16 + l16) * K;
            *((v8h*)&bf[t] + 0) = *(const v8h*)(brow + k0 + hh * 8);
            *((v8h*)&bf[t] + 1) = *(const v8h*)(brow + k0 + 16 + hh * 8);
        }
#pragma unroll
        for (int t = 0; t < 4; ++t)
            acc[0][t] = __builtin_amdgcn_wmma_f32_16x16x32_f16(
                false, a0, false, bf[t], (short)0, acc[0][t], false, false);
#pragma unroll
        for (int t = 0; t < 4; ++t)
            acc[1][t] = __builtin_amdgcn_wmma_f32_16x16x32_f16(
                false, a1, false, bf[t], (short)0, acc[1][t], false, false);
    }

    // ---- epilogue ----
    if (row0 + 32 <= M) {
        // uniform fast path: whole 32x64 tile in range, no predication
#pragma unroll
        for (int r = 0; r < 2; ++r) {
            const size_t rbase = (size_t)(row0 + r * 16 + hh * 8) * N;
#pragma unroll
            for (int t = 0; t < 4; ++t) {
                const int n  = col0 + t * 16 + l16;
                const float bn = bias[n];
                if (EPI == 1) {
                    _Float16* __restrict__ p = Ch + rbase + n;
#pragma unroll
                    for (int v = 0; v < 8; ++v)
                        p[(size_t)v * N] = (_Float16)fmaxf(acc[r][t][v] + bn, 0.0f);
                } else {
                    float* __restrict__ p = Cf + rbase + n;
#pragma unroll
                    for (int v = 0; v < 8; ++v)
                        p[(size_t)v * N] = acc[r][t][v] + bn;
                }
            }
        }
    } else {
        // tail row-block: per-row predication
#pragma unroll
        for (int r = 0; r < 2; ++r) {
#pragma unroll
            for (int t = 0; t < 4; ++t) {
                const int n  = col0 + t * 16 + l16;
                const float bn = bias[n];
#pragma unroll
                for (int v = 0; v < 8; ++v) {
                    const int mrow = row0 + r * 16 + hh * 8 + v;
                    if (mrow < M) {
                        float x = acc[r][t][v] + bn;
                        if (EPI == 1) Ch[(size_t)mrow * N + n] = (_Float16)fmaxf(x, 0.0f);
                        else          Cf[(size_t)mrow * N + n] = x;
                    }
                }
            }
        }
    }
}

// ---------------------------------------------------------------------------
// Softmax over 16 (levels*points) per (b,q,head).  16-lane shfl groups.
// total elements = B*Q*8*16, multiple of 256.
// ---------------------------------------------------------------------------
__global__ void k_softmax16(float* __restrict__ aw, int nelem) {
    int t = blockIdx.x * 256 + threadIdx.x;
    if (t >= nelem) return;
    float v = aw[t];
    float mx = v;
#pragma unroll
    for (int m = 1; m < 16; m <<= 1) mx = fmaxf(mx, __shfl_xor(mx, m, 32));
    float e = __expf(v - mx);
    float s = e;
#pragma unroll
    for (int m = 1; m < 16; m <<= 1) s += __shfl_xor(s, m, 32);
    aw[t] = e / s;
}

// ---------------------------------------------------------------------------
// Deformable attention core.  One wave per (b,q,head); lanes cover head_dim.
// val  : (B, QLEN, 256) f32     so : (B*Q, 256) f32     aw : (B*Q, 128) f32
// vr   : (B, 4, 2) valid_ratios attn: (B, Q, 256) f32
// ---------------------------------------------------------------------------
__global__ __launch_bounds__(256) void k_msda(const float* __restrict__ val,
                                              const float* __restrict__ so,
                                              const float* __restrict__ aw,
                                              const float* __restrict__ vr,
                                              float* __restrict__ attn) {
    int gid = blockIdx.x * 8 + (threadIdx.x >> 5);
    if (gid >= BATCH * QLEN * N_HEADS) return;
    const int lane = threadIdx.x & 31;
    const int h = gid & 7;
    const int q = (gid >> 3) % QLEN;
    const int b = gid / (QLEN * N_HEADS);

    // query's own level (levels are square: S = H = W)
    int lq, rem;
    if (q < 10000)      { lq = 0; rem = q; }
    else if (q < 12500) { lq = 1; rem = q - 10000; }
    else if (q < 13125) { lq = 2; rem = q - 12500; }
    else                { lq = 3; rem = q - 13125; }
    const int Sq = (lq == 0) ? 100 : (lq == 1) ? 50 : (lq == 2) ? 25 : 13;
    const int iy = rem / Sq;
    const int ix = rem % Sq;

    const float brx = (ix + 0.5f) / (vr[(b * 4 + lq) * 2 + 0] * Sq);
    const float bry = (iy + 0.5f) / (vr[(b * 4 + lq) * 2 + 1] * Sq);

    const size_t row = (size_t)(gid >> 3);            // b*QLEN + q
    const float* __restrict__ sop = so + row * 256 + h * 32;
    const float* __restrict__ awp = aw + row * 128 + h * 16;

    float acc = 0.0f;
#pragma unroll
    for (int l = 0; l < 4; ++l) {
        const int S    = (l == 0) ? 100 : (l == 1) ? 50 : (l == 2) ? 25 : 13;
        const int base = (l == 0) ? 0 : (l == 1) ? 10000 : (l == 2) ? 12500 : 13125;
        const float refx = brx * vr[(b * 4 + l) * 2 + 0];
        const float refy = bry * vr[(b * 4 + l) * 2 + 1];
#pragma unroll
        for (int p = 0; p < 4; ++p) {
            const float ox = sop[l * 8 + p * 2 + 0];
            const float oy = sop[l * 8 + p * 2 + 1];
            const float a  = awp[l * 4 + p];
            const float x  = (refx + ox / (float)S) * S - 0.5f;
            const float y  = (refy + oy / (float)S) * S - 0.5f;
            const float xf = floorf(x), yf = floorf(y);
            const float lx = x - xf, ly = y - yf;
            const int x0 = (int)xf, y0 = (int)yf;
            const float w[4] = {(1.f - ly) * (1.f - lx), (1.f - ly) * lx,
                                ly * (1.f - lx),          ly * lx};
#pragma unroll
            for (int t = 0; t < 4; ++t) {
                const int yi = y0 + (t >> 1), xi = x0 + (t & 1);
                const bool valid = (yi >= 0) & (yi < S) & (xi >= 0) & (xi < S);
                const int yc = min(max(yi, 0), S - 1);
                const int xc = min(max(xi, 0), S - 1);
                const float v = val[((size_t)b * QLEN + base + yc * S + xc) * 256 + h * 32 + lane];
                acc += valid ? a * w[t] * v : 0.0f;
            }
        }
    }
    attn[row * 256 + h * 32 + lane] = acc;
}

// ---------------------------------------------------------------------------
// Fused residual + LayerNorm over 256 channels.  Wave per row; 8 elems/lane.
// out <- LN(out + add) * scale + bias
// ---------------------------------------------------------------------------
__global__ __launch_bounds__(256) void k_resln(float* __restrict__ out,
                                               const float* __restrict__ add,
                                               const float* __restrict__ scale,
                                               const float* __restrict__ bias,
                                               int nrows) {
    int row = blockIdx.x * 8 + (threadIdx.x >> 5);
    if (row >= nrows) return;
    const int lane = threadIdx.x & 31;
    const size_t base = (size_t)row * 256 + lane * 8;
    float v[8];
    float s = 0.f, ss = 0.f;
#pragma unroll
    for (int i = 0; i < 8; ++i) {
        float x = out[base + i] + add[base + i];
        v[i] = x; s += x; ss += x * x;
    }
#pragma unroll
    for (int m = 1; m < 32; m <<= 1) {
        s  += __shfl_xor(s,  m, 32);
        ss += __shfl_xor(ss, m, 32);
    }
    const float mean = s * (1.0f / 256.0f);
    const float var  = ss * (1.0f / 256.0f) - mean * mean;
    const float inv  = rsqrtf(var + 1e-5f);
#pragma unroll
    for (int i = 0; i < 8; ++i) {
        const int c = lane * 8 + i;
        out[base + i] = (v[i] - mean) * inv * scale[c] + bias[c];
    }
}

// ---------------------------------------------------------------------------
extern "C" void kernel_launch(void* const* d_in, const int* in_sizes, int n_in,
                              void* d_out, int out_size, void* d_ws, size_t ws_size,
                              hipStream_t stream) {
    const float* src   = (const float*)d_in[0];
    const float* pos   = (const float*)d_in[1];
    const float* vr    = (const float*)d_in[2];
    const float* w_so  = (const float*)d_in[5];
    const float* b_so  = (const float*)d_in[6];
    const float* w_aw  = (const float*)d_in[7];
    const float* b_aw  = (const float*)d_in[8];
    const float* w_vp  = (const float*)d_in[9];
    const float* b_vp  = (const float*)d_in[10];
    const float* w_op  = (const float*)d_in[11];
    const float* b_op  = (const float*)d_in[12];
    const float* ln1_s = (const float*)d_in[13];
    const float* ln1_b = (const float*)d_in[14];
    const float* w_ff1 = (const float*)d_in[15];
    const float* b_ff1 = (const float*)d_in[16];
    const float* w_ff2 = (const float*)d_in[17];
    const float* b_ff2 = (const float*)d_in[18];
    const float* ln2_s = (const float*)d_in[19];
    const float* ln2_b = (const float*)d_in[20];
    (void)in_sizes; (void)n_in; (void)out_size; (void)ws_size;

    float* out = (float*)d_out;

    // workspace carve (all 256B-aligned)
    size_t woff = 0;
    auto carve = [&](size_t bytes) -> void* {
        void* p = (char*)d_ws + woff;
        woff += (bytes + 255) & ~(size_t)255;
        return p;
    };
    const size_t PER_LAYER_W16 = 65536ull * 3 + 32768 + 262144ull * 2;   // 753664 halves
    _Float16* W16  = (_Float16*)carve(PER_LAYER_W16 * NUM_LAYERS * 2);
    _Float16* q16  = (_Float16*)carve((size_t)MPAD * 256 * 2);
    _Float16* x16  = (_Float16*)carve((size_t)MPAD * 256 * 2);
    _Float16* h16  = (_Float16*)carve((size_t)MPAD * 1024 * 2);
    float*    valb = (float*)carve((size_t)MPAD * 256 * 4);
    float*    sob  = (float*)carve((size_t)MPAD * 256 * 4);
    float*    awb  = (float*)carve((size_t)MPAD * 128 * 4);
    float*    tmpb = (float*)carve((size_t)MPAD * 256 * 4);

    const int M   = MROWS;               // 26588
    const int NE  = MROWS * 256;         // activation elements
    const int GBX = (M + 127) / 128;     // 208 row-blocks of 128
    const dim3 BLK128(128), BLK256(256);

    // ---- one-time (per call) weight convert+transpose into f16 [N][K] ----
    for (int i = 0; i < NUM_LAYERS; ++i) {
        _Float16* Wl   = W16 + (size_t)i * PER_LAYER_W16;
        _Float16* vpT  = Wl;
        _Float16* soT  = vpT  + 65536;
        _Float16* awT  = soT  + 65536;
        _Float16* opT  = awT  + 32768;
        _Float16* ff1T = opT  + 65536;
        _Float16* ff2T = ff1T + 262144;
        k_transcvt<<<(65536 + 255) / 256, BLK256, 0, stream>>>(w_vp  + (size_t)i * 65536,  vpT,  256, 256);
        k_transcvt<<<(65536 + 255) / 256, BLK256, 0, stream>>>(w_so  + (size_t)i * 65536,  soT,  256, 256);
        k_transcvt<<<(32768 + 255) / 256, BLK256, 0, stream>>>(w_aw  + (size_t)i * 32768,  awT,  256, 128);
        k_transcvt<<<(65536 + 255) / 256, BLK256, 0, stream>>>(w_op  + (size_t)i * 65536,  opT,  256, 256);
        k_transcvt<<<(262144 + 255) / 256, BLK256, 0, stream>>>(w_ff1 + (size_t)i * 262144, ff1T, 256, 1024);
        k_transcvt<<<(262144 + 255) / 256, BLK256, 0, stream>>>(w_ff2 + (size_t)i * 262144, ff2T, 1024, 256);
    }

    // running state: out = src
    hipMemcpyAsync(out, src, (size_t)NE * sizeof(float), hipMemcpyDeviceToDevice, stream);

    for (int i = 0; i < NUM_LAYERS; ++i) {
        _Float16* Wl   = W16 + (size_t)i * PER_LAYER_W16;
        _Float16* vpT  = Wl;
        _Float16* soT  = vpT  + 65536;
        _Float16* awT  = soT  + 65536;
        _Float16* opT  = awT  + 32768;
        _Float16* ff1T = opT  + 65536;
        _Float16* ff2T = ff1T + 262144;

        // q16 = f16(out + pos), x16 = f16(out)
        k_addcvt<<<(NE + 255) / 256, BLK256, 0, stream>>>(out, pos, q16, x16, NE);

        // value / sampling-offset / attention-weight projections
        k_gemm<0><<<dim3(GBX, 4),  BLK128, 0, stream>>>(x16, vpT, b_vp + (size_t)i * 256, valb, nullptr, M, 256, 256);
        k_gemm<0><<<dim3(GBX, 4),  BLK128, 0, stream>>>(q16, soT, b_so + (size_t)i * 256, sob,  nullptr, M, 256, 256);
        k_gemm<0><<<dim3(GBX, 2),  BLK128, 0, stream>>>(q16, awT, b_aw + (size_t)i * 128, awb,  nullptr, M, 128, 256);

        // softmax over 16 (levels*points) per head
        k_softmax16<<<(M * 128 + 255) / 256, BLK256, 0, stream>>>(awb, M * 128);

        // deformable sampling core -> tmpb (B,Q,256)
        k_msda<<<(M * N_HEADS + 7) / 8, BLK256, 0, stream>>>(valb, sob, awb, vr, tmpb);

        // output projection: attn(f16) @ w_op
        k_cvt<<<(NE + 255) / 256, BLK256, 0, stream>>>(tmpb, q16, NE);
        k_gemm<0><<<dim3(GBX, 4),  BLK128, 0, stream>>>(q16, opT, b_op + (size_t)i * 256, tmpb, nullptr, M, 256, 256);

        // residual + LN1
        k_resln<<<(M + 7) / 8, BLK256, 0, stream>>>(out, tmpb, ln1_s + (size_t)i * 256, ln1_b + (size_t)i * 256, M);

        // FFN: relu(out @ w_ff1 + b) stored as f16, then @ w_ff2
        k_cvt<<<(NE + 255) / 256, BLK256, 0, stream>>>(out, x16, NE);
        k_gemm<1><<<dim3(GBX, 16), BLK128, 0, stream>>>(x16, ff1T, b_ff1 + (size_t)i * 1024, nullptr, h16, M, 1024, 256);
        k_gemm<0><<<dim3(GBX, 4),  BLK128, 0, stream>>>(h16, ff2T, b_ff2 + (size_t)i * 256,  tmpb, nullptr, M, 256, 1024);

        // residual + LN2
        k_resln<<<(M + 7) / 8, BLK256, 0, stream>>>(out, tmpb, ln2_s + (size_t)i * 256, ln2_b + (size_t)i * 256, M);
    }
}
